// FeatureEnhancerLayer_9268539424886
// MI455X (gfx1250) — matrix operand order repaired and checked by reading, not compiled, not run on hardware
//
#include <hip/hip_runtime.h>

#define DIM 768
#define HEADS 12
#define HD 64
#define BATCH 8
#define TTXT 256
#define TIMG 576

typedef __attribute__((ext_vector_type(16))) _Float16 v16h;
typedef __attribute__((ext_vector_type(8)))  _Float16 v8h;
typedef __attribute__((ext_vector_type(8)))  float    v8f;

// ---------------------------------------------------------------------------
// WMMA helpers (CDNA5 v_wmma_f32_16x16x32_f16, f32 accumulate)
// ---------------------------------------------------------------------------
__device__ __forceinline__ v8f wmma16x16x32(v16h a, v16h b, v8f c) {
  return __builtin_amdgcn_wmma_f32_16x16x32_f16(false, a, false, b, (short)0, c,
                                                false, false);
}

__device__ __forceinline__ v16h cat8(v8h lo, v8h hi) {
  return __builtin_shufflevector(lo, hi, 0, 1, 2, 3, 4, 5, 6, 7, 8, 9, 10, 11,
                                 12, 13, 14, 15);
}

// A-matrix 16x32 f16 fragment from row-major [m][k] source (ISA layout:
// lane<16 row m=lane, K = {0..7, 16..23}+8*kg) -> two contiguous 16B loads.
__device__ __forceinline__ v16h load_afrag(const _Float16* base, int ld) {
  int lane = threadIdx.x & 31;
  int m = lane & 15, kg = lane >> 4;
  const _Float16* p = base + m * ld + 8 * kg;
  v8h lo = *(const v8h*)p;
  v8h hi = *(const v8h*)(p + 16);
  return cat8(lo, hi);
}

// B-matrix 32x16 f16 fragment where the SOURCE is stored K-contiguous per
// N-row ([n][k], stride ld): lane holds n=lane%16, K = {0..7, 16..23}+8*kg.
__device__ __forceinline__ v16h load_bfrag_t(const _Float16* base, int ld) {
  int lane = threadIdx.x & 31;
  int n = lane & 15, kg = lane >> 4;
  const _Float16* p = base + n * ld + 8 * kg;
  v8h lo = *(const v8h*)p;
  v8h hi = *(const v8h*)(p + 16);
  return cat8(lo, hi);
}

// Async global -> LDS 16-byte copy (CDNA5, tracked by ASYNCcnt).
__device__ __forceinline__ void async_cp16(const _Float16* gsrc, _Float16* ldst) {
  unsigned lds_off = (unsigned)(unsigned long long)ldst;
  unsigned long long ga = (unsigned long long)(const void*)gsrc;
  asm volatile("global_load_async_to_lds_b128 %0, %1, off"
               :
               : "v"(lds_off), "v"(ga)
               : "memory");
}

// ---------------------------------------------------------------------------
// Tiled WMMA GEMM: C[M,768] = A[M,768](f16) x W^T[768n][768k](f16) + bias.
// Block = 128 threads (4 waves), 64x64 output tile, K-step 32.
// Double-buffered LDS tiles filled by async global->LDS DMA; tile t+1 streams
// in while WMMAs consume tile t.
// ---------------------------------------------------------------------------
__global__ void __launch_bounds__(128)
gemm_f16_kernel(const _Float16* __restrict__ A, const _Float16* __restrict__ Bt,
                const float* __restrict__ bias, _Float16* __restrict__ C16,
                float* __restrict__ C32, int M, int relu) {
  const int N = DIM, K = DIM;
  __shared__ __align__(16) _Float16 sA[2][64 * 32];   // [m][k]
  __shared__ __align__(16) _Float16 sBt[2][64 * 32];  // [n][k]
  int mbase = blockIdx.y * 64, nbase = blockIdx.x * 64;
  int wave = threadIdx.x >> 5, lane = threadIdx.x & 31;

  auto issue_tile = [&](int k0, int buf) {
#pragma unroll
    for (int u = 0; u < 2; ++u) {
      int idx = threadIdx.x * 8 + u * 1024;  // 2048 halfs per tile
      int r = idx >> 5, cc = idx & 31;
      async_cp16(A + (size_t)(mbase + r) * K + (k0 + cc), &sA[buf][idx]);
      async_cp16(Bt + (size_t)(nbase + r) * K + (k0 + cc), &sBt[buf][idx]);
    }
  };

  v8f acc[4];
#pragma unroll
  for (int c = 0; c < 4; ++c)
#pragma unroll
    for (int r = 0; r < 8; ++r) acc[c][r] = 0.0f;

  issue_tile(0, 0);
  const int NT = K / 32;  // 24
  for (int it = 0; it < NT; ++it) {
    int buf = it & 1;
    asm volatile("s_wait_asynccnt 0" ::: "memory");
    __syncthreads();  // tile `it` resident; all waves done reading buf^1
    if (it + 1 < NT) issue_tile((it + 1) * 32, buf ^ 1);

    v16h a = load_afrag(&sA[buf][wave * 16 * 32], 32);
    v16h b0 = load_bfrag_t(&sBt[buf][0 * 16 * 32], 32);
    v16h b1 = load_bfrag_t(&sBt[buf][1 * 16 * 32], 32);
    v16h b2 = load_bfrag_t(&sBt[buf][2 * 16 * 32], 32);
    v16h b3 = load_bfrag_t(&sBt[buf][3 * 16 * 32], 32);
    acc[0] = wmma16x16x32(a, b0, acc[0]);
    acc[1] = wmma16x16x32(a, b1, acc[1]);
    acc[2] = wmma16x16x32(a, b2, acc[2]);
    acc[3] = wmma16x16x32(a, b3, acc[3]);
    __syncthreads();  // done reading buf before it is refilled at it+2
  }

  int g = lane >> 4, col = lane & 15;
  int mrow = mbase + wave * 16;
#pragma unroll
  for (int c = 0; c < 4; ++c) {
    int n = nbase + c * 16 + col;
    float bv = bias ? bias[n] : 0.0f;
#pragma unroll
    for (int r = 0; r < 8; ++r) {
      int m = mrow + r + 8 * g;
      float v = acc[c][r] + bv;
      if (relu) v = fmaxf(v, 0.0f);
      if (C16) C16[(size_t)m * N + n] = (_Float16)v;
      if (C32) C32[(size_t)m * N + n] = v;
    }
  }
}

// ---------------------------------------------------------------------------
// Flash-style attention: grid (Sq/64, HEADS, BATCH), 128 threads (4 waves).
// K for (b,h) staged in LDS via async global->LDS b128 (ASYNCcnt);
// V staged transposed as sVt[e][key] so P*V B-fragments are contiguous.
// Each wave owns 16 query rows; online softmax; scores and P*V via WMMA.
// Q/K/V layout: [b, s, h*64+e], row stride DIM.
// ---------------------------------------------------------------------------
__global__ void __launch_bounds__(128)
attn_kernel(const _Float16* __restrict__ Q, const _Float16* __restrict__ Kf,
            const _Float16* __restrict__ Vf, _Float16* __restrict__ O,
            int Sq, int Skv) {
  extern __shared__ _Float16 smem[];
  _Float16* sK = smem;                         // [key][64]
  _Float16* sVt = smem + (size_t)Skv * HD;     // [64][Skv]
  _Float16* sP = smem + 2 * (size_t)Skv * HD;  // 4 waves x 16x32
  int b = blockIdx.z, h = blockIdx.y;
  int qbase = blockIdx.x * 64;
  int wave = threadIdx.x >> 5, lane = threadIdx.x & 31;
  int g = lane >> 4, col = lane & 15;

  const _Float16* gK = Kf + ((size_t)b * Skv) * DIM + h * HD;
  const _Float16* gV = Vf + ((size_t)b * Skv) * DIM + h * HD;

  // K: async global->LDS DMA, row-major [key][64].
  for (int t = threadIdx.x; t < Skv * 8; t += 128) {
    int r = t >> 3, c = (t & 7) * 8;
    async_cp16(gK + (size_t)r * DIM + c, &sK[r * HD + c]);
  }
  // V: load rows, scatter-transpose into sVt[e][key].
  for (int t = threadIdx.x; t < Skv * 8; t += 128) {
    int r = t >> 3, c = (t & 7) * 8;
    uint4 d = *(const uint4*)(gV + (size_t)r * DIM + c);
    const _Float16* h8 = (const _Float16*)&d;
#pragma unroll
    for (int j = 0; j < 8; ++j) sVt[(size_t)(c + j) * Skv + r] = h8[j];
  }
  asm volatile("s_wait_asynccnt 0" ::: "memory");
  __syncthreads();

  const _Float16* gQ = Q + ((size_t)b * Sq + qbase + wave * 16) * DIM + h * HD;
  v16h qf0 = load_afrag(gQ, DIM);
  v16h qf1 = load_afrag(gQ + 32, DIM);

  float m_r[8], l_r[8];
  v8f o[4];
#pragma unroll
  for (int r = 0; r < 8; ++r) { m_r[r] = -1e30f; l_r[r] = 0.0f; }
#pragma unroll
  for (int c = 0; c < 4; ++c)
#pragma unroll
    for (int r = 0; r < 8; ++r) o[c][r] = 0.0f;

  const float scale = 0.125f;  // 1/sqrt(64)
  _Float16* myP = sP + wave * (16 * 32);

  for (int kt = 0; kt < Skv; kt += 32) {
    // scores: Q(16x64) x K^T(64x32); batch fragment loads, then WMMAs.
    v16h kb00 = load_bfrag_t(sK + (size_t)kt * HD, HD);
    v16h kb01 = load_bfrag_t(sK + (size_t)kt * HD + 32, HD);
    v16h kb10 = load_bfrag_t(sK + (size_t)(kt + 16) * HD, HD);
    v16h kb11 = load_bfrag_t(sK + (size_t)(kt + 16) * HD + 32, HD);
    v8f s0, s1;
#pragma unroll
    for (int r = 0; r < 8; ++r) { s0[r] = 0.0f; s1[r] = 0.0f; }
    s0 = wmma16x16x32(qf0, kb00, s0);
    s0 = wmma16x16x32(qf1, kb01, s0);
    s1 = wmma16x16x32(qf0, kb10, s1);
    s1 = wmma16x16x32(qf1, kb11, s1);

#pragma unroll
    for (int r = 0; r < 8; ++r) {
      s0[r] *= scale; s1[r] *= scale;
      float mx = fmaxf(s0[r], s1[r]);
#pragma unroll
      for (int d = 1; d < 16; d <<= 1) mx = fmaxf(mx, __shfl_xor(mx, d, 32));
      float mn = fmaxf(m_r[r], mx);
      float alpha = __expf(m_r[r] - mn);
      m_r[r] = mn;
      s0[r] = __expf(s0[r] - mn);
      s1[r] = __expf(s1[r] - mn);
      float rs = s0[r] + s1[r];
#pragma unroll
      for (int d = 1; d < 16; d <<= 1) rs += __shfl_xor(rs, d, 32);
      l_r[r] = l_r[r] * alpha + rs;
#pragma unroll
      for (int c = 0; c < 4; ++c) o[c][r] *= alpha;
    }

    // Transpose P (C-layout -> A-layout) through per-wave LDS staging.
#pragma unroll
    for (int r = 0; r < 8; ++r) {
      myP[(r + 8 * g) * 32 + col] = (_Float16)s0[r];
      myP[(r + 8 * g) * 32 + col + 16] = (_Float16)s1[r];
    }
    asm volatile("s_wait_dscnt 0" ::: "memory");  // wave-local LDS RAW
    v16h pf = load_afrag(myP, 32);
    v16h vb0 = load_bfrag_t(sVt + (size_t)(0 * 16) * Skv + kt, Skv);
    v16h vb1 = load_bfrag_t(sVt + (size_t)(1 * 16) * Skv + kt, Skv);
    v16h vb2 = load_bfrag_t(sVt + (size_t)(2 * 16) * Skv + kt, Skv);
    v16h vb3 = load_bfrag_t(sVt + (size_t)(3 * 16) * Skv + kt, Skv);
    o[0] = wmma16x16x32(pf, vb0, o[0]);
    o[1] = wmma16x16x32(pf, vb1, o[1]);
    o[2] = wmma16x16x32(pf, vb2, o[2]);
    o[3] = wmma16x16x32(pf, vb3, o[3]);
  }

  _Float16* gO = O + ((size_t)b * Sq + qbase + wave * 16) * DIM + h * HD;
#pragma unroll
  for (int c = 0; c < 4; ++c)
#pragma unroll
    for (int r = 0; r < 8; ++r)
      gO[(size_t)(r + 8 * g) * DIM + c * 16 + col] = (_Float16)(o[c][r] / l_r[r]);
}

// ---------------------------------------------------------------------------
// LayerNorm over last dim (768). One 256-thread block per row.
// orow remap lets the t2i stream land directly in the zero-padded layout.
// ---------------------------------------------------------------------------
__global__ void __launch_bounds__(256)
layernorm_kernel(const _Float16* __restrict__ x, const float* __restrict__ gam,
                 const float* __restrict__ bet, _Float16* __restrict__ y,
                 int rin, int rout) {
  int row = blockIdx.x;
  int tid = threadIdx.x;
  const _Float16* xr = x + (size_t)row * DIM;
  float v[3];
#pragma unroll
  for (int i = 0; i < 3; ++i) v[i] = (float)xr[tid + i * 256];
  float s = v[0] + v[1] + v[2];
  float s2 = v[0] * v[0] + v[1] * v[1] + v[2] * v[2];
#pragma unroll
  for (int d = 1; d < 32; d <<= 1) {
    s += __shfl_xor(s, d, 32);
    s2 += __shfl_xor(s2, d, 32);
  }
  __shared__ float ls[8], ls2[8];
  int w = tid >> 5, ln = tid & 31;
  if (ln == 0) { ls[w] = s; ls2[w] = s2; }
  __syncthreads();
  float ts = 0.f, ts2 = 0.f;
#pragma unroll
  for (int i = 0; i < 8; ++i) { ts += ls[i]; ts2 += ls2[i]; }
  float mu = ts * (1.0f / DIM);
  float var = ts2 * (1.0f / DIM) - mu * mu;
  float rs = rsqrtf(var + 1e-5f);
  int orow = (row / rin) * rout + (row % rin);
  _Float16* yr = y + (size_t)orow * DIM;
#pragma unroll
  for (int i = 0; i < 3; ++i) {
    int c = tid + i * 256;
    yr[c] = (_Float16)(((v[i] - mu) * rs) * gam[c] + bet[c]);
  }
}

// ---------------------------------------------------------------------------
// Conversion / fill utilities
// ---------------------------------------------------------------------------
__global__ void cvt_f32_f16_kernel(const float* __restrict__ in,
                                   _Float16* __restrict__ out, int n) {
  int i = blockIdx.x * 256 + threadIdx.x;
  if (i < n) out[i] = (_Float16)in[i];
}

// (HEADS, DIM, HD) f32 -> transposed [n = h*64+e][k = d] f16
__global__ void cvt_qkvw_T_kernel(const float* __restrict__ in,
                                  _Float16* __restrict__ out) {
  int i = blockIdx.x * 256 + threadIdx.x;
  if (i < DIM * DIM) {
    int n = i / DIM, d = i % DIM;
    int h = n / HD, e = n % HD;
    out[i] = (_Float16)in[((size_t)h * DIM + d) * HD + e];
  }
}

// [k][n] f32 row-major -> transposed [n][k] f16
__global__ void cvt_w_T_kernel(const float* __restrict__ in,
                               _Float16* __restrict__ out) {
  int i = blockIdx.x * 256 + threadIdx.x;
  if (i < DIM * DIM) {
    int n = i / DIM, k = i % DIM;
    out[i] = (_Float16)in[(size_t)k * DIM + n];
  }
}

__global__ void zero_f16_kernel(_Float16* p, int n) {
  int i = blockIdx.x * 256 + threadIdx.x;
  if (i < n) p[i] = (_Float16)0.0f;
}

// ---------------------------------------------------------------------------
// Host orchestration
// ---------------------------------------------------------------------------
extern "C" void kernel_launch(void* const* d_in, const int* in_sizes, int n_in,
                              void* d_out, int out_size, void* d_ws, size_t ws_size,
                              hipStream_t stream) {
  (void)in_sizes; (void)n_in; (void)out_size; (void)ws_size;
  auto F = [&](int i) { return (const float*)d_in[i]; };
  const int MT = BATCH * TTXT;  // 2048
  const int MI = BATCH * TIMG;  // 4608

  _Float16* base = (_Float16*)d_ws;
  size_t off = 0;
  auto alloc = [&](size_t n) {
    _Float16* p = base + off;
    off += (n + 63) & ~(size_t)63;
    return p;
  };
  const size_t WN = (size_t)DIM * DIM;

  _Float16 *w_q[4], *w_k[4], *w_v[4];
  for (int s = 0; s < 4; ++s) { w_q[s] = alloc(WN); w_k[s] = alloc(WN); w_v[s] = alloc(WN); }
  _Float16* w_ffn[4];  // text w1, text w2, image w1, image w2 (all transposed)
  for (int s = 0; s < 4; ++s) w_ffn[s] = alloc(WN);
  _Float16* Xt  = alloc((size_t)MT * DIM);
  _Float16* Xi  = alloc((size_t)MI * DIM);
  _Float16* Qb  = alloc((size_t)MI * DIM);
  _Float16* Kb  = alloc((size_t)MI * DIM);
  _Float16* Vb  = alloc((size_t)MI * DIM);
  _Float16* AOb = alloc((size_t)MI * DIM);
  _Float16* Txt = alloc((size_t)MT * DIM);
  _Float16* Img = alloc((size_t)MI * DIM);
  _Float16* T2I = alloc((size_t)MI * DIM);
  _Float16* I2T = alloc((size_t)MI * DIM);
  _Float16* TxtO = alloc((size_t)MI * DIM);
  _Float16* ImgO = alloc((size_t)MI * DIM);
  _Float16* Hh   = alloc((size_t)MI * DIM);

  // Input dict order: 0 text, 1 image, then sat/sai/cti/cit each
  // {wq,bq,wk,bk,wv,bv,g,b}, ffn_text {w1,b1,w2,b2}, ffn_image {...},
  // norm_text {g,b}, norm_image {g,b}.
  int mha_base[4] = {2, 10, 18, 26};

  dim3 cb(256);
  int wgrid = (DIM * DIM + 255) / 256;
  for (int s = 0; s < 4; ++s) {
    cvt_qkvw_T_kernel<<<wgrid, cb, 0, stream>>>(F(mha_base[s] + 0), w_q[s]);
    cvt_qkvw_T_kernel<<<wgrid, cb, 0, stream>>>(F(mha_base[s] + 2), w_k[s]);
    cvt_qkvw_T_kernel<<<wgrid, cb, 0, stream>>>(F(mha_base[s] + 4), w_v[s]);
  }
  cvt_w_T_kernel<<<wgrid, cb, 0, stream>>>(F(34), w_ffn[0]);
  cvt_w_T_kernel<<<wgrid, cb, 0, stream>>>(F(36), w_ffn[1]);
  cvt_w_T_kernel<<<wgrid, cb, 0, stream>>>(F(38), w_ffn[2]);
  cvt_w_T_kernel<<<wgrid, cb, 0, stream>>>(F(40), w_ffn[3]);
  cvt_f32_f16_kernel<<<(MT * DIM + 255) / 256, cb, 0, stream>>>(F(0), Xt, MT * DIM);
  cvt_f32_f16_kernel<<<(MI * DIM + 255) / 256, cb, 0, stream>>>(F(1), Xi, MI * DIM);

  auto gemm = [&](const _Float16* A, const _Float16* Bw, const float* bias,
                  _Float16* C16, float* C32, int M, int relu) {
    dim3 gg(DIM / 64, M / 64);
    gemm_f16_kernel<<<gg, dim3(128), 0, stream>>>(A, Bw, bias, C16, C32, M, relu);
  };
  auto attn = [&](const _Float16* Qp, const _Float16* Kp, const _Float16* Vp,
                  _Float16* Op, int Sq, int Skv) {
    dim3 gg(Sq / 64, HEADS, BATCH);
    size_t sm = ((size_t)2 * Skv * HD + 4 * 16 * 32) * sizeof(_Float16);
    attn_kernel<<<gg, dim3(128), sm, stream>>>(Qp, Kp, Vp, Op, Sq, Skv);
  };
  auto lnorm = [&](const _Float16* x, const float* g, const float* bb, _Float16* y,
                   int rows, int rin, int rout) {
    layernorm_kernel<<<rows, dim3(256), 0, stream>>>(x, g, bb, y, rin, rout);
  };

  // Stage 1: text self-attention (sat)
  gemm(Xt, w_q[0], F(3), Qb, nullptr, MT, 0);
  gemm(Xt, w_k[0], F(5), Kb, nullptr, MT, 0);
  gemm(Xt, w_v[0], F(7), Vb, nullptr, MT, 0);
  attn(Qb, Kb, Vb, AOb, TTXT, TTXT);
  lnorm(AOb, F(8), F(9), Txt, MT, TTXT, TTXT);

  // Stage 2: image self-attention (sai)
  gemm(Xi, w_q[1], F(11), Qb, nullptr, MI, 0);
  gemm(Xi, w_k[1], F(13), Kb, nullptr, MI, 0);
  gemm(Xi, w_v[1], F(15), Vb, nullptr, MI, 0);
  attn(Qb, Kb, Vb, AOb, TIMG, TIMG);
  lnorm(AOb, F(16), F(17), Img, MI, TIMG, TIMG);

  // Stage 3: cti (Q from txt, K/V from img) -> t2i, written into padded layout
  gemm(Txt, w_q[2], F(19), Qb, nullptr, MT, 0);
  gemm(Img, w_k[2], F(21), Kb, nullptr, MI, 0);
  gemm(Img, w_v[2], F(23), Vb, nullptr, MI, 0);
  attn(Qb, Kb, Vb, AOb, TTXT, TIMG);
  zero_f16_kernel<<<(MI * DIM + 255) / 256, cb, 0, stream>>>(T2I, MI * DIM);
  lnorm(AOb, F(24), F(25), T2I, MT, TTXT, TIMG);

  // Stage 4: cit (Q from img, K/V from txt) -> i2t
  gemm(Img, w_q[3], F(27), Qb, nullptr, MI, 0);
  gemm(Txt, w_k[3], F(29), Kb, nullptr, MT, 0);
  gemm(Txt, w_v[3], F(31), Vb, nullptr, MT, 0);
  attn(Qb, Kb, Vb, AOb, TIMG, TTXT);
  lnorm(AOb, F(32), F(33), I2T, MI, TIMG, TIMG);

  // Stage 5: final norms (streams swapped per reference)
  lnorm(I2T, F(42), F(43), TxtO, MI, TIMG, TIMG);  // norm_text(i2t)
  lnorm(T2I, F(44), F(45), ImgO, MI, TIMG, TIMG);  // norm_image(t2i padded)

  // Stage 6: FFNs -> f32 outputs (concatenated: ffn_text, then ffn_image)
  float* out = (float*)d_out;
  gemm(TxtO, w_ffn[0], F(35), Hh, nullptr, MI, 1);
  gemm(Hh, w_ffn[1], F(37), nullptr, out, MI, 0);
  gemm(ImgO, w_ffn[2], F(39), Hh, nullptr, MI, 1);
  gemm(Hh, w_ffn[3], F(41), nullptr, out + (size_t)MI * DIM, MI, 0);
}
